// SS2D_25031069401735
// MI455X (gfx1250) — compile-verified, matrix operand order checked by reading, not run on hardware
//
#include <hip/hip_runtime.h>
#include <hip/hip_bf16.h>

// ---------------------------------------------------------------------------
// SS2D (VMamba selective-scan 2D) for gfx1250.
// B=1, H=W=64, L=4096, D_MODEL=96, D_INNER=192, D_STATE=16, DT_RANK=6, K=4.
// Projections use v_wmma_f32_16x16x32_bf16 (f32 accumulate); scan is 768
// independent lane-serial recurrences with 16 states in VGPRs.
// ---------------------------------------------------------------------------

#define Lsz    4096
#define HW     64
#define DM     96
#define DI     192
#define DS     16
#define DTR    6
#define KD     4
#define CROWS  38   // DT_RANK + 2*D_STATE

typedef __attribute__((ext_vector_type(16))) __bf16 v16bf;
typedef __attribute__((ext_vector_type(8)))  float  v8f;

// workspace layout (float offsets)
#define OFF_XI    ((size_t)0)          // 192*4096  (later reused as ynt [m][192])
#define OFF_XC    ((size_t)786432)     // 192*4096  (later reused as yg  [d][4096])
#define OFF_XDBL  ((size_t)1572864)    // 4*38*4096
#define OFF_YBUF  ((size_t)2195456)    // 768*4096
#define OFF_SM    ((size_t)5341184)    // small vectors below
// small sub-offsets (relative to OFF_SM)
#define SM_S      0        // 192
#define SM_G      192      // 192
#define SM_B      384      // 192
#define SM_D      576      // 768
#define SM_A      1344     // 12288

// float -> bf16 via hardware conversion (v_cvt_*bf16* on gfx1250)
__device__ __forceinline__ __bf16 f2bf(float f) { return (__bf16)f; }

// K-index packing for 16-bit A/B fragments (ISA 7.12.2, 16x32 bf16)
__device__ __forceinline__ int kmap(int j, bool hi) {
  int v = j >> 1;
  int base = (v < 4) ? ((hi ? 8 : 0) + 2 * v)
                     : (16 + (hi ? 8 : 0) + 2 * (v - 4));
  return base + (j & 1);
}

// scan orders: 0=row-major, 1=transpose, 2=rev(row), 3=rev(transpose)
__device__ __forceinline__ int order_k(int k, int l) {
  switch (k) {
    case 0:  return l;
    case 1:  return ((l & 63) << 6) | (l >> 6);
    case 2:  return (Lsz - 1) - l;
    default: { int j = (Lsz - 1) - l; return ((j & 63) << 6) | (j >> 6); }
  }
}

// ---------------------------------------------------------------------------
// Kernel 0: rep_g = mean(rep); derive s, gamma, beta, Ds, A = -exp(...)
// ---------------------------------------------------------------------------
__global__ void k_precompute(const float* __restrict__ rep,
                             const float* __restrict__ smw, const float* __restrict__ smb,
                             const float* __restrict__ alw, const float* __restrict__ alb,
                             const float* __restrict__ dsw, const float* __restrict__ dsb,
                             const float* __restrict__ gw,  const float* __restrict__ gb,
                             const float* __restrict__ bw,  const float* __restrict__ bb,
                             float* __restrict__ sm) {
  __shared__ float part[256];
  __shared__ float rg[64];
  int t = threadIdx.x;
  // rep mean: 4 threads per channel, each sums a contiguous 1024-elem chunk
  {
    int r = t >> 2, q = t & 3;
    const float* p = rep + (size_t)r * Lsz + q * (Lsz / 4);
    float s = 0.f;
    #pragma unroll 4
    for (int l = 0; l < Lsz / 4; ++l) s += p[l];
    part[t] = s;
  }
  __syncthreads();
  if ((t & 3) == 0)
    rg[t >> 2] = (part[t] + part[t + 1] + part[t + 2] + part[t + 3]) *
                 (1.0f / (float)Lsz);
  __syncthreads();
  if (t < DI) {
    float as = smb[t], ag = gb[t], ab = bb[t];
    #pragma unroll 8
    for (int j = 0; j < 64; ++j) {
      float r = rg[j];
      as += r * smw[t * 64 + j];
      ag += r * gw[t * 64 + j];
      ab += r * bw[t * 64 + j];
    }
    sm[SM_S + t] = 1.0f + as;
    sm[SM_G + t] = ag;
    sm[SM_B + t] = ab;
  }
  for (int i = t; i < KD * DI; i += 256) {
    float a = dsb[i];
    #pragma unroll 8
    for (int j = 0; j < 64; ++j) a += rg[j] * dsw[i * 64 + j];
    sm[SM_D + i] = a;
  }
  for (int i = t; i < KD * DI * DS; i += 256) {
    float a = alb[i];
    #pragma unroll 8
    for (int j = 0; j < 64; ++j) a += rg[j] * alw[i * 64 + j];
    sm[SM_A + i] = -__expf(a);
  }
}

// ---------------------------------------------------------------------------
// Kernel 1: xi_s[d][l] = s[d] * (x @ in_proj_w^T)   via bf16 WMMA
// 256 l-tiles x 12 d-tiles = 3072 waves
// ---------------------------------------------------------------------------
__global__ void k_inproj(const float* __restrict__ x, const float* __restrict__ W,
                         const float* __restrict__ sm, float* __restrict__ xi) {
  int wid  = (blockIdx.x * blockDim.x + threadIdx.x) >> 5;
  int lane = threadIdx.x & 31;
  int lt = wid & 255;
  int dt = wid >> 8;
  bool hi = lane >= 16;
  int mn = lane & 15;
  v8f c = {};
  for (int kb = 0; kb < DM; kb += 32) {
    v16bf a, b;
    #pragma unroll
    for (int j = 0; j < 16; ++j) {
      int k = kb + kmap(j, hi);
      a[j] = f2bf(x[(size_t)(lt * 16 + mn) * DM + k]);      // A: M=l, K=c
      b[j] = f2bf(W[(size_t)(dt * 16 + mn) * DM + k]);      // B: N=d, K=c
    }
    c = __builtin_amdgcn_wmma_f32_16x16x32_bf16(false, a, false, b,
                                                (short)0, c, false, false);
  }
  int d = dt * 16 + mn;
  float sc = sm[SM_S + d];
  #pragma unroll
  for (int j = 0; j < 8; ++j) {
    int m = j + (hi ? 8 : 0);
    xi[(size_t)d * Lsz + lt * 16 + m] = c[j] * sc;
  }
}

// ---------------------------------------------------------------------------
// Kernel 2: depthwise 3x3 conv (SAME, zero pad) + bias + SiLU
// ---------------------------------------------------------------------------
__global__ void k_conv(const float* __restrict__ xi, const float* __restrict__ w,
                       const float* __restrict__ b, float* __restrict__ xc) {
  int gid = blockIdx.x * blockDim.x + threadIdx.x;
  int d = gid / Lsz, l = gid % Lsz;
  int h = l >> 6, wq = l & 63;
  float acc = b[d];
  const float* wd = w + d * 9;
  const float* xd = xi + (size_t)d * Lsz;
  #pragma unroll
  for (int dy = -1; dy <= 1; ++dy) {
    int hh = h + dy;
    if ((unsigned)hh >= HW) continue;
    #pragma unroll
    for (int dx = -1; dx <= 1; ++dx) {
      int ww = wq + dx;
      if ((unsigned)ww >= HW) continue;
      acc += xd[hh * HW + ww] * wd[(dy + 1) * 3 + (dx + 1)];
    }
  }
  xc[gid] = acc / (1.0f + __expf(-acc));   // silu
}

// ---------------------------------------------------------------------------
// Kernel 3: x_dbl[k][c][l] = sum_d W[k][c][d] * xc[d][order_k(l)]  via WMMA
// 4k x 3 c-tiles x 256 l-tiles = 3072 waves (permutation folded into B loads)
// ---------------------------------------------------------------------------
__global__ void k_xdbl(const float* __restrict__ xpw, const float* __restrict__ xc,
                       float* __restrict__ xdbl) {
  int wid  = (blockIdx.x * blockDim.x + threadIdx.x) >> 5;
  int lane = threadIdx.x & 31;
  int k4 = wid / 768;
  int rem = wid % 768;
  int ct = rem >> 8;        // 0..2  (c tiles, 38 rows padded to 48)
  int lt = rem & 255;       // l tiles
  bool hi = lane >= 16;
  int mn = lane & 15;
  int lphys = order_k(k4, lt * 16 + mn);   // per-lane B column physical index
  v8f c = {};
  for (int db = 0; db < DI; db += 32) {
    v16bf a, b;
    #pragma unroll
    for (int j = 0; j < 16; ++j) {
      int d = db + kmap(j, hi);
      int crow = ct * 16 + mn;
      a[j] = (crow < CROWS) ? f2bf(xpw[((size_t)k4 * CROWS + crow) * DI + d])
                            : f2bf(0.0f);
      b[j] = f2bf(xc[(size_t)d * Lsz + lphys]);
    }
    c = __builtin_amdgcn_wmma_f32_16x16x32_bf16(false, a, false, b,
                                                (short)0, c, false, false);
  }
  #pragma unroll
  for (int j = 0; j < 8; ++j) {
    int crow = ct * 16 + j + (hi ? 8 : 0);
    if (crow < CROWS)
      xdbl[((size_t)k4 * CROWS + crow) * Lsz + lt * 16 + mn] = c[j];
  }
}

// ---------------------------------------------------------------------------
// Kernel 4: selective scan. One lane per (k,d); 16 states in VGPRs.
// grid = 4 (k), block = 192 (d). dt-projection + softplus fused.
// ---------------------------------------------------------------------------
__global__ void k_scan(const float* __restrict__ xc, const float* __restrict__ xdbl,
                       const float* __restrict__ dtw_all, const float* __restrict__ dtb_all,
                       const float* __restrict__ sm, float* __restrict__ ybuf) {
  int k4 = blockIdx.x;
  int d  = threadIdx.x;
  float A[DS], h[DS], dtw[DTR];
  #pragma unroll
  for (int n = 0; n < DS; ++n) {
    A[n] = sm[SM_A + ((size_t)k4 * DI + d) * DS + n];
    h[n] = 0.f;
  }
  #pragma unroll
  for (int r = 0; r < DTR; ++r) dtw[r] = dtw_all[((size_t)k4 * DI + d) * DTR + r];
  float dtb = dtb_all[k4 * DI + d];
  float Dv  = sm[SM_D + k4 * DI + d];
  const float* xdk = xdbl + (size_t)k4 * CROWS * Lsz;
  float* yo = ybuf + ((size_t)k4 * DI + d) * Lsz;
  const float* xcd = xc + (size_t)d * Lsz;

  for (int l = 0; l < Lsz; ++l) {
    int lp = order_k(k4, l);
    float u = xcd[lp];
    __builtin_prefetch(&xdk[l + 16], 0, 1);        // global_prefetch_b8
    float dt = dtb;
    #pragma unroll
    for (int r = 0; r < DTR; ++r) dt += dtw[r] * xdk[(size_t)r * Lsz + l];
    float delta = (dt > 20.f) ? dt : log1pf(__expf(dt));   // softplus
    float du = delta * u;
    float y = 0.f;
    #pragma unroll
    for (int n = 0; n < DS; ++n) {
      float Bn = xdk[(size_t)(DTR + n) * Lsz + l];
      float Cn = xdk[(size_t)(DTR + DS + n) * Lsz + l];
      h[n] = __expf(delta * A[n]) * h[n] + du * Bn;
      y += h[n] * Cn;
    }
    yo[l] = y + Dv * u;
  }
}

// ---------------------------------------------------------------------------
// Kernel 5: gather inverse perms + instance norm + affine.
// grid = 192 (d), block = 256. Writes ynt[m][192] (row-major m) for out_proj.
// ---------------------------------------------------------------------------
__global__ void k_norm(const float* __restrict__ ybuf, const float* __restrict__ sm,
                       float* __restrict__ yg, float* __restrict__ ynt) {
  int d = blockIdx.x, t = threadIdx.x;
  __shared__ float red[256], red2[256];
  float s = 0.f, s2 = 0.f;
  const float* y0 = ybuf + ((size_t)0 * DI + d) * Lsz;
  const float* y1 = ybuf + ((size_t)1 * DI + d) * Lsz;
  const float* y2 = ybuf + ((size_t)2 * DI + d) * Lsz;
  const float* y3 = ybuf + ((size_t)3 * DI + d) * Lsz;
  for (int m = t; m < Lsz; m += 256) {
    int sw = ((m & 63) << 6) | (m >> 6);
    float g = y0[m] + y1[sw] + y2[(Lsz - 1) - m] + y3[(Lsz - 1) - sw];
    yg[(size_t)d * Lsz + m] = g;
    s += g; s2 += g * g;
  }
  red[t] = s; red2[t] = s2;
  __syncthreads();
  for (int o = 128; o > 0; o >>= 1) {
    if (t < o) { red[t] += red[t + o]; red2[t] += red2[t + o]; }
    __syncthreads();
  }
  float mu   = red[0] * (1.0f / (float)Lsz);
  float var  = red2[0] * (1.0f / (float)Lsz) - mu * mu;
  float rstd = rsqrtf(var + 1e-5f);
  float ga = sm[SM_G + d], be = sm[SM_B + d];
  for (int m = t; m < Lsz; m += 256)
    ynt[(size_t)m * DI + d] = (yg[(size_t)d * Lsz + m] - mu) * rstd * ga + be;
}

// ---------------------------------------------------------------------------
// Kernel 6: out[m][c] = ynt[m][:] @ out_proj_w[c][:]  via WMMA
// 256 l-tiles x 6 c-tiles = 1536 waves
// ---------------------------------------------------------------------------
__global__ void k_outproj(const float* __restrict__ ynt, const float* __restrict__ W,
                          float* __restrict__ out) {
  int wid  = (blockIdx.x * blockDim.x + threadIdx.x) >> 5;
  int lane = threadIdx.x & 31;
  int lt = wid & 255;
  int nt = wid >> 8;        // 0..5
  bool hi = lane >= 16;
  int mn = lane & 15;
  v8f c = {};
  for (int kb = 0; kb < DI; kb += 32) {
    v16bf a, b;
    #pragma unroll
    for (int j = 0; j < 16; ++j) {
      int k = kb + kmap(j, hi);
      a[j] = f2bf(ynt[(size_t)(lt * 16 + mn) * DI + k]);   // A: M=l, K=d
      b[j] = f2bf(W[(size_t)(nt * 16 + mn) * DI + k]);     // B: N=c, K=d
    }
    c = __builtin_amdgcn_wmma_f32_16x16x32_bf16(false, a, false, b,
                                                (short)0, c, false, false);
  }
  #pragma unroll
  for (int j = 0; j < 8; ++j) {
    int m = j + (hi ? 8 : 0);
    out[(size_t)(lt * 16 + m) * DM + nt * 16 + mn] = c[j];
  }
}

// ---------------------------------------------------------------------------
extern "C" void kernel_launch(void* const* d_in, const int* in_sizes, int n_in,
                              void* d_out, int out_size, void* d_ws, size_t ws_size,
                              hipStream_t stream) {
  (void)in_sizes; (void)n_in; (void)out_size; (void)ws_size;
  const float* x   = (const float*)d_in[0];
  const float* rep = (const float*)d_in[1];
  const float* inw = (const float*)d_in[2];
  const float* scw = (const float*)d_in[3];
  const float* scb = (const float*)d_in[4];
  const float* smw = (const float*)d_in[5];
  const float* smb = (const float*)d_in[6];
  const float* xpw = (const float*)d_in[7];
  const float* dtw = (const float*)d_in[8];
  const float* dtb = (const float*)d_in[9];
  const float* alw = (const float*)d_in[10];
  const float* alb = (const float*)d_in[11];
  const float* dsw = (const float*)d_in[12];
  const float* dsb = (const float*)d_in[13];
  const float* gw  = (const float*)d_in[14];
  const float* gb  = (const float*)d_in[15];
  const float* bw  = (const float*)d_in[16];
  const float* bb  = (const float*)d_in[17];
  const float* ow  = (const float*)d_in[18];

  float* ws  = (float*)d_ws;
  float* out = (float*)d_out;

  float* xi   = ws + OFF_XI;     // xi*s, [d][l]; later reused as ynt [m][d]
  float* xc   = ws + OFF_XC;     // conv out, [d][l]; later reused as yg [d][m]
  float* xdbl = ws + OFF_XDBL;   // [k][38][l]
  float* ybuf = ws + OFF_YBUF;   // [k][d][l] (scan index)
  float* sm   = ws + OFF_SM;     // small derived vectors

  k_precompute<<<1, 256, 0, stream>>>(rep, smw, smb, alw, alb, dsw, dsb,
                                      gw, gb, bw, bb, sm);
  k_inproj <<<768, 128, 0, stream>>>(x, inw, sm, xi);
  k_conv   <<<(DI * Lsz) / 256, 256, 0, stream>>>(xi, scw, scb, xc);
  k_xdbl   <<<768, 128, 0, stream>>>(xpw, xc, xdbl);
  k_scan   <<<KD, DI, 0, stream>>>(xc, xdbl, dtw, dtb, sm, ybuf);
  k_norm   <<<DI, 256, 0, stream>>>(ybuf, sm, xc /*yg*/, xi /*ynt*/);
  k_outproj<<<384, 128, 0, stream>>>(xi /*ynt*/, ow, out);
}
